// EntropyComponent_9019431322128
// MI455X (gfx1250) — matrix operand order repaired and checked by reading, not compiled, not run on hardware
//
#include <hip/hip_runtime.h>
#include <math.h>

// ---------------------------------------------------------------------------
// Types for CDNA5 WMMA (wave32): A/B = 16 x bf16, C/D = 8 x f32
// ---------------------------------------------------------------------------
typedef __attribute__((ext_vector_type(16))) __bf16 bf16x16;
typedef __attribute__((ext_vector_type(8)))  __bf16 bf16x8;
typedef __attribute__((ext_vector_type(8)))  float  f32x8;

static __device__ __forceinline__ float siluf(float x) {
  return x / (1.f + __expf(-x));
}
static __device__ __forceinline__ float geluf(float x) {
  return 0.5f * x * (1.f + erff(x * 0.70710678118654752f));
}
static __device__ __forceinline__ float softplusf(float x) {
  return (x > 20.f) ? x : log1pf(expf(x));
}

// Block-wide sum over 256 threads (8 waves of 32).
static __device__ __forceinline__ float blockReduceSum(float v, float* red) {
#pragma unroll
  for (int off = 16; off > 0; off >>= 1) v += __shfl_xor(v, off, 32);
  const int w = threadIdx.x >> 5, lane = threadIdx.x & 31;
  if (lane == 0) red[w] = v;
  __syncthreads();
  float r = (lane < 8) ? red[lane] : 0.f;
#pragma unroll
  for (int off = 4; off > 0; off >>= 1) r += __shfl_xor(r, off, 32);
  r = __shfl(r, 0, 32);           // broadcast within wave
  __syncthreads();                // allow 'red' reuse by a later call
  return r;
}

// ---------------------------------------------------------------------------
// 1. fp32 -> bf16 copy
// ---------------------------------------------------------------------------
__global__ void k_f32_to_bf16(const float* __restrict__ in,
                              __bf16* __restrict__ out, int n) {
  int i = blockIdx.x * blockDim.x + threadIdx.x;
  if (i < n) out[i] = (__bf16)in[i];
}

// ---------------------------------------------------------------------------
// 2. weight transpose+convert: W[K,N] fp32 -> Wt[N,K] bf16
// ---------------------------------------------------------------------------
__global__ void k_transpose_bf16(const float* __restrict__ W,
                                 __bf16* __restrict__ Wt, int K, int N) {
  int i = blockIdx.x * blockDim.x + threadIdx.x;
  if (i >= K * N) return;
  int nn = i / K, kk = i % K;
  Wt[i] = (__bf16)W[(size_t)kk * N + nn];
}

// ---------------------------------------------------------------------------
// 3. WMMA GEMM:  out[M,N] = A[M,K](bf16) x Wt[N,K]^T(bf16) + bias, act
//    One 16x16 D-tile per wave; 8 waves/block cover 128 columns.
//    Fragment fill uses contiguous k order (k = 16*kh + s) for A(16x32) and B(32x16):
//      A: lane(m=l&15, kh=l>>4): elems = A[m][k+16*kh .. +15]
//      B: lane(n=l&15, kh):      elems = W [k+16*kh .. +15][n]  (= Wt[n][k+16*kh..])
//      D: lane(n=l&15, kh), vgpr r -> row m = r + 8*kh
// ---------------------------------------------------------------------------
__global__ __launch_bounds__(256)
void k_gemm_wmma(const __bf16* __restrict__ A, const __bf16* __restrict__ Wt,
                 const float* __restrict__ bias, float* __restrict__ out,
                 __bf16* __restrict__ out_bf, int M, int N, int K, int act) {
  const int wave = threadIdx.x >> 5;
  const int lane = threadIdx.x & 31;
  const int tn = blockIdx.x * 128 + wave * 16;
  const int tm = blockIdx.y * 16;
  if (tn >= N) return;                       // wave-uniform exit
  const int half = lane >> 4;
  const int idx  = lane & 15;
  const __bf16* arow = A  + (size_t)(tm + idx) * K;
  const __bf16* brow = Wt + (size_t)(tn + idx) * K;
  f32x8 c = {0.f, 0.f, 0.f, 0.f, 0.f, 0.f, 0.f, 0.f};
  for (int k = 0; k < K; k += 32) {
    __builtin_prefetch(arow + k + 256, 0, 1);   // global_prefetch_b8
    __builtin_prefetch(brow + k + 256, 0, 1);
    bf16x8 a0 = *(const bf16x8*)(arow + k + 16 * half);
    bf16x8 a1 = *(const bf16x8*)(arow + k + 16 * half + 8);
    bf16x16 b = *(const bf16x16*)(brow + k + 16 * half);
    bf16x16 a;
#pragma unroll
    for (int j = 0; j < 8; ++j) { a[j] = a0[j]; a[j + 8] = a1[j]; }
    c = __builtin_amdgcn_wmma_f32_16x16x32_bf16(
        /*neg_a=*/false, a, /*neg_b=*/false, b,
        /*c_mod=*/(short)0, c, /*reuse_a=*/false, /*reuse_b=*/false);
  }
  const int n = tn + idx;
  const float bs = bias ? bias[n] : 0.f;
#pragma unroll
  for (int r = 0; r < 8; ++r) {
    float v = c[r] + bs;
    if (act == 1) v = geluf(v);
    const size_t o = (size_t)(tm + r + 8 * half) * N + n;
    out[o] = v;
    if (out_bf) out_bf[o] = (__bf16)v;
  }
}

// ---------------------------------------------------------------------------
// 4. causal depthwise conv (k=4) over xBC channels of zxbcdt, + bias + SiLU
//    zx: [B*L, 2208] (xBC at cols 1024..2175) -> convb: [B*L, 1152]
// ---------------------------------------------------------------------------
__global__ void k_dwconv_silu(const float* __restrict__ zx,
                              const float* __restrict__ cw,   // [1152,4]
                              const float* __restrict__ cb,   // [1152]
                              float* __restrict__ out, int L, int n) {
  int i = blockIdx.x * blockDim.x + threadIdx.x;
  if (i >= n) return;
  int c = i % 1152;
  int r = i / 1152;           // global row b*L + l
  int l = r % L;
  float s = cb[c];
#pragma unroll
  for (int t = 0; t < 4; ++t) {
    int li = l - 3 + t;
    if (li >= 0) s += zx[(size_t)(r + li - l) * 2208 + 1024 + c] * cw[c * 4 + t];
  }
  out[(size_t)r * 1152 + c] = siluf(s);
}

// ---------------------------------------------------------------------------
// 5. selective scan: one block per (b, head). State h[32][64] lives in
//    registers (8 floats/thread). LDS stages per-step B/C/x vectors.
// ---------------------------------------------------------------------------
__global__ __launch_bounds__(256)
void k_scan(const float* __restrict__ conv,   // [B*L,1152] x|B|C
            const float* __restrict__ zx,     // [B*L,2208] dt at 2176+
            const float* __restrict__ dt_bias, const float* __restrict__ A_log,
            const float* __restrict__ Dp, float* __restrict__ y, int L) {
  const int b = blockIdx.x >> 5;
  const int head = blockIdx.x & 31;
  const int tid = threadIdx.x;
  const int p = tid >> 3, g = tid & 7, n0 = g * 8;
  __shared__ float Bs[64], Cs[64], xs[32], sdt[2];
  const float Ah  = -expf(A_log[head]);
  const float dtb = dt_bias[head];
  const float Dh  = Dp[head];
  float st[8];
#pragma unroll
  for (int i = 0; i < 8; ++i) st[i] = 0.f;
  const float* cb = conv + (size_t)b * L * 1152;
  const float* zb = zx + (size_t)b * L * 2208;
  float* yb = y + (size_t)b * L * 1024;
  for (int t = 0; t < L; ++t) {
    const float* crow = cb + (size_t)t * 1152;
    if (tid < 64)        Bs[tid] = crow[1024 + tid];
    else if (tid < 128)  Cs[tid - 64] = crow[1088 + (tid - 64)];
    else if (tid < 160)  xs[tid - 128] = crow[head * 32 + (tid - 128)];
    else if (tid == 160) {
      float dtv = softplusf(zb[(size_t)t * 2208 + 2176 + head] + dtb);
      sdt[0] = dtv; sdt[1] = expf(dtv * Ah);
    }
    __syncthreads();
    const float dtv = sdt[0], dA = sdt[1];
    const float coef = dtv * xs[p];
    float acc = 0.f;
#pragma unroll
    for (int i = 0; i < 8; ++i) {
      st[i] = st[i] * dA + coef * Bs[n0 + i];
      acc += st[i] * Cs[n0 + i];
    }
    acc += __shfl_xor(acc, 1, 32);
    acc += __shfl_xor(acc, 2, 32);
    acc += __shfl_xor(acc, 4, 32);
    if (g == 0) yb[(size_t)t * 1024 + head * 32 + p] = acc + Dh * xs[p];
    __syncthreads();
  }
}

// ---------------------------------------------------------------------------
// 6. gated RMSNorm: out_bf = rms(y * silu(z), w); one block per row (1024)
// ---------------------------------------------------------------------------
__global__ __launch_bounds__(256)
void k_gated_rms(const float* __restrict__ Y, const float* __restrict__ ZX,
                 const float* __restrict__ w, __bf16* __restrict__ out_bf) {
  __shared__ float red[8];
  const size_t row = blockIdx.x;
  const float* y = Y + row * 1024;
  const float* z = ZX + row * 2208;     // z = cols 0..1023
  float g[4]; float ss = 0.f;
#pragma unroll
  for (int i = 0; i < 4; ++i) {
    int c = threadIdx.x + 256 * i;
    float gv = y[c] * siluf(z[c]);
    g[i] = gv; ss += gv * gv;
  }
  ss = blockReduceSum(ss, red);
  float scale = rsqrtf(ss * (1.f / 1024.f) + 1e-5f);
#pragma unroll
  for (int i = 0; i < 4; ++i) {
    int c = threadIdx.x + 256 * i;
    out_bf[row * 1024 + c] = (__bf16)(g[i] * scale * w[c]);
  }
}

// ---------------------------------------------------------------------------
// 7. residual add + RMSNorm (row = 512): out = rms(a+b, w); fp32 + bf16 out
// ---------------------------------------------------------------------------
__global__ __launch_bounds__(256)
void k_add_rms(const float* __restrict__ A, const float* __restrict__ Bv,
               const float* __restrict__ w, float* __restrict__ out,
               __bf16* __restrict__ out_bf) {
  __shared__ float red[8];
  const size_t row = blockIdx.x;
  const float* a = A + row * 512;
  const float* b = Bv + row * 512;
  float v[2]; float ss = 0.f;
#pragma unroll
  for (int i = 0; i < 2; ++i) {
    int c = threadIdx.x + 256 * i;
    float t = a[c] + b[c];
    v[i] = t; ss += t * t;
  }
  ss = blockReduceSum(ss, red);
  float scale = rsqrtf(ss * (1.f / 512.f) + 1e-5f);
#pragma unroll
  for (int i = 0; i < 2; ++i) {
    int c = threadIdx.x + 256 * i;
    float o = v[i] * scale * w[c];
    out[row * 512 + c] = o;
    if (out_bf) out_bf[row * 512 + c] = (__bf16)o;
  }
}

// ---------------------------------------------------------------------------
// 8. LayerNorm (row = 512) with optional residual: out = ln(x (+ r)) * w + b
// ---------------------------------------------------------------------------
__global__ __launch_bounds__(256)
void k_layernorm(const float* __restrict__ X, const float* __restrict__ R,
                 const float* __restrict__ w, const float* __restrict__ bia,
                 float* __restrict__ out, __bf16* __restrict__ out_bf) {
  __shared__ float red[8];
  const size_t row = blockIdx.x;
  const float* x = X + row * 512;
  float v[2]; float s = 0.f, sq = 0.f;
#pragma unroll
  for (int i = 0; i < 2; ++i) {
    int c = threadIdx.x + 256 * i;
    float t = x[c];
    if (R) t += R[row * 512 + c];
    v[i] = t; s += t; sq += t * t;
  }
  s  = blockReduceSum(s, red);
  sq = blockReduceSum(sq, red);
  float mean = s * (1.f / 512.f);
  float var  = sq * (1.f / 512.f) - mean * mean;
  float inv  = rsqrtf(var + 1e-5f);
#pragma unroll
  for (int i = 0; i < 2; ++i) {
    int c = threadIdx.x + 256 * i;
    float o = (v[i] - mean) * inv * w[c] + bia[c];
    out[row * 512 + c] = o;
    if (out_bf) out_bf[row * 512 + c] = (__bf16)o;
  }
}

// ---------------------------------------------------------------------------
// 9. im2col for strided down-conv (k=3, s=2, pad=1): x3[B,2048,512] ->
//    col[(b*1024+lo), i*3+t] (bf16).  K ordering matches down_w [O][I][T].
// ---------------------------------------------------------------------------
__global__ void k_im2col(const float* __restrict__ x3, __bf16* __restrict__ col,
                         int n) {
  int i = blockIdx.x * blockDim.x + threadIdx.x;
  if (i >= n) return;
  int m = i / 1536, k = i % 1536;
  int ci = k / 3, t = k % 3;
  int b = m / 1024, lo = m % 1024;
  int li = 2 * lo + t - 1;
  float val = (li >= 0 && li < 2048) ? x3[((size_t)b * 2048 + li) * 512 + ci] : 0.f;
  col[i] = (__bf16)val;
}

// ---------------------------------------------------------------------------
// 10. fused attention, one block per (b,h); 8 waves; per-wave score row in LDS
//     Q,K,V,O: [B*1024, 512] fp32, head h at cols h*64..h*64+63. scale 1/8.
// ---------------------------------------------------------------------------
__global__ __launch_bounds__(256)
void k_attention(const float* __restrict__ Q, const float* __restrict__ K,
                 const float* __restrict__ V, float* __restrict__ O) {
  const int b = blockIdx.x >> 3;
  const int h = blockIdx.x & 7;
  const int wave = threadIdx.x >> 5;
  const int lane = threadIdx.x & 31;
  const int L = 1024;
  __shared__ float sc[8][1024];
  __shared__ float qv[8][64];
  const float* Qb = Q + (size_t)b * L * 512 + h * 64;
  const float* Kb = K + (size_t)b * L * 512 + h * 64;
  const float* Vb = V + (size_t)b * L * 512 + h * 64;
  float* Ob = O + (size_t)b * L * 512 + h * 64;
  for (int q = wave; q < L; q += 8) {
    qv[wave][lane]      = Qb[(size_t)q * 512 + lane];
    qv[wave][32 + lane] = Qb[(size_t)q * 512 + 32 + lane];
    __builtin_amdgcn_wave_barrier();
    // pass 1: scores + running max (lane-strided over keys)
    float mx = -1e30f;
    for (int kk = lane; kk < L; kk += 32) {
      const float* kr = Kb + (size_t)kk * 512;
      float s = 0.f;
#pragma unroll
      for (int d = 0; d < 64; ++d) s += qv[wave][d] * kr[d];
      s *= 0.125f;
      sc[wave][kk] = s;
      mx = fmaxf(mx, s);
    }
#pragma unroll
    for (int off = 16; off > 0; off >>= 1) mx = fmaxf(mx, __shfl_xor(mx, off, 32));
    __builtin_amdgcn_wave_barrier();
    float sum = 0.f;
    for (int kk = lane; kk < L; kk += 32) {
      float e = expf(sc[wave][kk] - mx);
      sc[wave][kk] = e;
      sum += e;
    }
#pragma unroll
    for (int off = 16; off > 0; off >>= 1) sum += __shfl_xor(sum, off, 32);
    __builtin_amdgcn_wave_barrier();
    // pass 2: weighted sum of V; lane owns dims 2*lane, 2*lane+1
    float a0 = 0.f, a1 = 0.f;
    for (int kk = 0; kk < L; ++kk) {
      float pr = sc[wave][kk];                 // LDS broadcast
      const float* vr = Vb + (size_t)kk * 512;
      a0 += pr * vr[2 * lane];
      a1 += pr * vr[2 * lane + 1];
    }
    float inv = 1.f / sum;
    Ob[(size_t)q * 512 + 2 * lane]     = a0 * inv;
    Ob[(size_t)q * 512 + 2 * lane + 1] = a1 * inv;
    __builtin_amdgcn_wave_barrier();
  }
}

// ===========================================================================
// host orchestration
// ===========================================================================
extern "C" void kernel_launch(void* const* d_in, const int* in_sizes, int n_in,
                              void* d_out, int out_size, void* d_ws, size_t ws_size,
                              hipStream_t stream) {
  (void)in_sizes; (void)n_in; (void)out_size; (void)ws_size;
  // --- inputs (setup_inputs order) -----------------------------------------
  const float* x      = (const float*)d_in[0];    // [2,2048,1024]
  const float* w_in   = (const float*)d_in[1];    // [1024,512]
  const float* b_in   = (const float*)d_in[2];
  // m1
  const float* in_proj1 = (const float*)d_in[3];  // [512,2208]
  const float* conv_w1  = (const float*)d_in[4];  // [1152,4]
  const float* conv_b1  = (const float*)d_in[5];
  const float* dt_b1    = (const float*)d_in[6];
  const float* A_log1   = (const float*)d_in[7];
  const float* D1       = (const float*)d_in[8];
  const float* mnw1     = (const float*)d_in[9];
  const float* out_p1   = (const float*)d_in[10]; // [1024,512]
  // m2
  const float* in_proj2 = (const float*)d_in[11];
  const float* conv_w2  = (const float*)d_in[12];
  const float* conv_b2  = (const float*)d_in[13];
  const float* dt_b2    = (const float*)d_in[14];
  const float* A_log2   = (const float*)d_in[15];
  const float* D2       = (const float*)d_in[16];
  const float* mnw2     = (const float*)d_in[17];
  const float* out_p2   = (const float*)d_in[18];
  const float* norm1  = (const float*)d_in[19];
  const float* norm2  = (const float*)d_in[20];
  const float* down_w = (const float*)d_in[21];   // [512,512,3] = [N=512,K=1536]
  const float* down_b = (const float*)d_in[22];
  const float* wq = (const float*)d_in[23]; const float* bq = (const float*)d_in[24];
  const float* wk = (const float*)d_in[25]; const float* bk = (const float*)d_in[26];
  const float* wv = (const float*)d_in[27]; const float* bv = (const float*)d_in[28];
  const float* wo = (const float*)d_in[29]; const float* bo = (const float*)d_in[30];
  const float* ln1w = (const float*)d_in[31]; const float* ln1b = (const float*)d_in[32];
  const float* ln2w = (const float*)d_in[33]; const float* ln2b = (const float*)d_in[34];
  const float* w1 = (const float*)d_in[35]; const float* b1 = (const float*)d_in[36];
  const float* w2 = (const float*)d_in[37]; const float* b2 = (const float*)d_in[38];
  const float* lnw = (const float*)d_in[39]; const float* lnb = (const float*)d_in[40];
  float* outp = (float*)d_out;

  const int Lm = 2048;               // mamba sequence length
  const int M1 = 2 * Lm;             // 4096 rows in mamba phase
  const int M2 = 2 * 1024;           // 2048 rows in transformer phase

  // --- workspace layout ----------------------------------------------------
  char* base = (char*)d_ws;
  size_t pos = 0;
  auto alloc = [&](size_t bytes) -> void* {
    void* p = base + pos;
    pos += (bytes + 255) & ~(size_t)255;
    return p;
  };
  // persistent: bf16 weights
  __bf16* w_inT = (__bf16*)alloc((size_t)512 * 1024 * 2);
  __bf16* in1T  = (__bf16*)alloc((size_t)2208 * 512 * 2);
  __bf16* out1T = (__bf16*)alloc((size_t)512 * 1024 * 2);
  __bf16* in2T  = (__bf16*)alloc((size_t)2208 * 512 * 2);
  __bf16* out2T = (__bf16*)alloc((size_t)512 * 1024 * 2);
  __bf16* wqT   = (__bf16*)alloc((size_t)512 * 512 * 2);
  __bf16* wkT   = (__bf16*)alloc((size_t)512 * 512 * 2);
  __bf16* wvT   = (__bf16*)alloc((size_t)512 * 512 * 2);
  __bf16* woT   = (__bf16*)alloc((size_t)512 * 512 * 2);
  __bf16* w1T   = (__bf16*)alloc((size_t)1024 * 512 * 2);
  __bf16* w2T   = (__bf16*)alloc((size_t)512 * 1024 * 2);
  __bf16* downT = (__bf16*)alloc((size_t)512 * 1536 * 2);
  // persistent activations across phases
  __bf16* xbf  = (__bf16*)alloc((size_t)M1 * 1024 * 2);
  float*  x1   = (float*)alloc((size_t)M1 * 512 * 4);
  __bf16* x1bf = (__bf16*)alloc((size_t)M1 * 512 * 2);
  float*  x2   = (float*)alloc((size_t)M1 * 512 * 4);
  __bf16* x2bf = (__bf16*)alloc((size_t)M1 * 512 * 2);
  float*  x3   = (float*)alloc((size_t)M1 * 512 * 4);
  // scratch region (mamba layout and transformer layout overlap)
  const size_t scratch0 = pos;
  float*  zx    = (float*)alloc((size_t)M1 * 2208 * 4);
  float*  convb = (float*)alloc((size_t)M1 * 1152 * 4);
  float*  ybuf  = (float*)alloc((size_t)M1 * 1024 * 4);
  __bf16* ybf   = (__bf16*)alloc((size_t)M1 * 1024 * 2);
  float*  mo    = (float*)alloc((size_t)M1 * 512 * 4);
  pos = scratch0;                         // transformer phase reuses scratch
  __bf16* colbf  = (__bf16*)alloc((size_t)M2 * 1536 * 2);
  float*  xd     = (float*)alloc((size_t)M2 * 512 * 4);
  __bf16* xdbf   = (__bf16*)alloc((size_t)M2 * 512 * 2);
  float*  qf     = (float*)alloc((size_t)M2 * 512 * 4);
  float*  kf     = (float*)alloc((size_t)M2 * 512 * 4);
  float*  vf     = (float*)alloc((size_t)M2 * 512 * 4);
  float*  attn   = (float*)alloc((size_t)M2 * 512 * 4);
  __bf16* attnbf = (__bf16*)alloc((size_t)M2 * 512 * 2);
  float*  of     = (float*)alloc((size_t)M2 * 512 * 4);
  float*  xa     = (float*)alloc((size_t)M2 * 512 * 4);
  __bf16* xabf   = (__bf16*)alloc((size_t)M2 * 512 * 2);
  float*  h1     = (float*)alloc((size_t)M2 * 1024 * 4);
  __bf16* h1bf   = (__bf16*)alloc((size_t)M2 * 1024 * 2);
  float*  ff     = (float*)alloc((size_t)M2 * 512 * 4);
  float*  xb     = (float*)alloc((size_t)M2 * 512 * 4);

  // --- helpers -------------------------------------------------------------
  auto cvt = [&](const float* src, __bf16* dst, int n) {
    k_f32_to_bf16<<<(n + 255) / 256, 256, 0, stream>>>(src, dst, n);
  };
  auto trans = [&](const float* W, __bf16* Wt, int K, int N) {
    k_transpose_bf16<<<(K * N + 255) / 256, 256, 0, stream>>>(W, Wt, K, N);
  };
  auto gemm = [&](const __bf16* A, const __bf16* Wt, const float* bias,
                  float* out, __bf16* obf, int M, int N, int K, int act) {
    dim3 g((N + 127) / 128, M / 16);
    k_gemm_wmma<<<g, 256, 0, stream>>>(A, Wt, bias, out, obf, M, N, K, act);
  };

  // --- weight conversion (every call; deterministic, cheap) ----------------
  trans(w_in, w_inT, 1024, 512);
  trans(in_proj1, in1T, 512, 2208);
  trans(out_p1, out1T, 1024, 512);
  trans(in_proj2, in2T, 512, 2208);
  trans(out_p2, out2T, 1024, 512);
  trans(wq, wqT, 512, 512); trans(wk, wkT, 512, 512);
  trans(wv, wvT, 512, 512); trans(wo, woT, 512, 512);
  trans(w1, w1T, 512, 1024); trans(w2, w2T, 1024, 512);
  cvt(down_w, downT, 512 * 1536);            // already [O][I][T] = [N][K]

  // --- stem: x @ w_in + b_in ----------------------------------------------
  cvt(x, xbf, M1 * 1024);
  gemm(xbf, w_inT, b_in, x1, x1bf, M1, 512, 1024, 0);

  // --- mamba blocks --------------------------------------------------------
  auto run_mamba = [&](const __bf16* xin_bf, const float* resid,
                       const __bf16* inT, const __bf16* outT,
                       const float* cw, const float* cb2, const float* dtb,
                       const float* alog, const float* Dv, const float* mnw,
                       const float* post_w, float* xout, __bf16* xout_bf) {
    gemm(xin_bf, inT, nullptr, zx, nullptr, M1, 2208, 512, 0);
    k_dwconv_silu<<<(M1 * 1152 + 255) / 256, 256, 0, stream>>>(
        zx, cw, cb2, convb, Lm, M1 * 1152);
    k_scan<<<64, 256, 0, stream>>>(convb, zx, dtb, alog, Dv, ybuf, Lm);
    k_gated_rms<<<M1, 256, 0, stream>>>(ybuf, zx, mnw, ybf);
    gemm(ybf, outT, nullptr, mo, nullptr, M1, 512, 1024, 0);
    k_add_rms<<<M1, 256, 0, stream>>>(mo, resid, post_w, xout, xout_bf);
  };
  run_mamba(x1bf, x1, in1T, out1T, conv_w1, conv_b1, dt_b1, A_log1, D1, mnw1,
            norm1, x2, x2bf);
  run_mamba(x2bf, x2, in2T, out2T, conv_w2, conv_b2, dt_b2, A_log2, D2, mnw2,
            norm2, x3, nullptr);

  // --- downsample conv (im2col + GEMM) ------------------------------------
  k_im2col<<<(M2 * 1536 + 255) / 256, 256, 0, stream>>>(x3, colbf, M2 * 1536);
  gemm(colbf, downT, down_b, xd, xdbf, M2, 512, 1536, 0);

  // --- transformer layer ---------------------------------------------------
  gemm(xdbf, wqT, bq, qf, nullptr, M2, 512, 512, 0);
  gemm(xdbf, wkT, bk, kf, nullptr, M2, 512, 512, 0);
  gemm(xdbf, wvT, bv, vf, nullptr, M2, 512, 512, 0);
  k_attention<<<16, 256, 0, stream>>>(qf, kf, vf, attn);
  cvt(attn, attnbf, M2 * 512);
  gemm(attnbf, woT, bo, of, nullptr, M2, 512, 512, 0);
  k_layernorm<<<M2, 256, 0, stream>>>(xd, of, ln1w, ln1b, xa, xabf);
  gemm(xabf, w1T, b1, h1, h1bf, M2, 1024, 512, 1);       // + GELU
  gemm(h1bf, w2T, b2, ff, nullptr, M2, 512, 1024, 0);
  k_layernorm<<<M2, 256, 0, stream>>>(xa, ff, ln2w, ln2b, xb, nullptr);
  k_layernorm<<<M2, 256, 0, stream>>>(xb, nullptr, lnw, lnb, outp, nullptr);
}
